// WindowedAttentionHead_6219112644957
// MI455X (gfx1250) — matrix operand order repaired and checked by reading, not compiled, not run on hardware
//
#include <hip/hip_runtime.h>

// CDNA5 / gfx1250, wave32. Windowed shifted attention, one window per WG.
// B=32, W=8192, C=64, ws=128, shift=64, nw=64 -> 2048 windows.
// HBM-bound (~400 MB traffic vs 8.8 GFLOP): f16 WMMA w/ f32 accum, NT cache
// hints on all streaming global traffic.

typedef __attribute__((ext_vector_type(16))) _Float16 v16h;
typedef __attribute__((ext_vector_type(8)))  _Float16 v8h;
typedef __attribute__((ext_vector_type(8)))  float    v8f;
typedef __attribute__((ext_vector_type(4)))  float    v4f;

#define WSZ    128
#define SHIFT  64
#define HD     64
#define WTOT   8192
#define NW     (WTOT / WSZ)
#define QKSCALE 0.125f  // 64^-0.5

union Frag16 { v16h f; v8h h[2]; };

// Build a v16h WMMA operand from two 16-byte LDS chunks.
__device__ __forceinline__ v16h make_frag(const _Float16* p0, const _Float16* p1) {
    Frag16 u;
    u.h[0] = *(const v8h*)p0;
    u.h[1] = *(const v8h*)p1;
    return u.f;
}

__global__ __launch_bounds__(256) void swin_attn_fwd(
    const float* __restrict__ q, const float* __restrict__ k,
    const float* __restrict__ v, const float* __restrict__ table,
    float* __restrict__ xout, float* __restrict__ attn_out)
{
    // LDS layout (50176 B):
    //   [0,16K)   sQ   128x64 f16   \__ aliased by sP 128x128 f16 after GEMM1
    //   [16K,32K) sK   128x64 f16   /
    //   [32K,48K) sVt  64x128 f16 (channel-major: V transposed)
    //   [48K, +1020) sBias 255 f32
    __shared__ __align__(16) char smem[50176];
    _Float16* sQ    = (_Float16*)(smem);
    _Float16* sK    = (_Float16*)(smem + 16384);
    _Float16* sP    = (_Float16*)(smem);
    _Float16* sVt   = (_Float16*)(smem + 32768);
    float*    sBias = (float*)   (smem + 49152);

    const int widx = blockIdx.x;
    const int b    = widx / NW;
    const int w    = widx % NW;
    const int tid  = threadIdx.x;
    const int lane = tid & 31;
    const int wv   = tid >> 5;

    // ---------------- load + cyclic shift (roll -64) + f32->f16 ----------------
    // Thread t covers half of one window row (32 channels). Streaming reads ->
    // non-temporal (each (b,row) is consumed by exactly one workgroup).
    {
        const int row  = tid >> 1;
        const int c0   = (tid & 1) * 32;
        const int rowg = (w * WSZ + row + SHIFT) & (WTOT - 1);
        const size_t g = ((size_t)b * WTOT + rowg) * HD + c0;
        const v4f* sq = (const v4f*)(q + g);
        const v4f* sk = (const v4f*)(k + g);
        const v4f* sv = (const v4f*)(v + g);
        _Float16* dq = sQ + row * HD + c0;
        _Float16* dk = sK + row * HD + c0;
#pragma unroll
        for (int i = 0; i < 8; ++i) {
            v4f fq = __builtin_nontemporal_load(sq + i);
            v4f fk = __builtin_nontemporal_load(sk + i);
            v4f fv = __builtin_nontemporal_load(sv + i);
            const int cc = c0 + 4 * i;
#pragma unroll
            for (int j = 0; j < 4; ++j) {
                dq[4*i+j] = (_Float16)fq[j];
                dk[4*i+j] = (_Float16)fk[j];
                sVt[(cc+j)*WSZ + row] = (_Float16)fv[j];  // transpose V
            }
        }
        if (tid < 2 * WSZ - 1) sBias[tid] = table[tid];
    }
    __syncthreads();

    // Per-lane fragment geometry (ISA 7.12.2, 16-bit operands, wave32):
    //   A: lane L holds row (L%16); lanes 0-15 -> K chunks {0-7,16-23}, lanes 16-31 -> {8-15,24-31}
    //   B: lane L holds col (L%16); lanes 0-15 -> K chunks {0-7,16-23}, lanes 16-31 -> {8-15,24-31}
    //   C/D: VGPR m, lanes 0-15 -> (M=m, N=lane); lanes 16-31 -> (M=m+8, N=lane-16)
    const int rl     = lane & 15;
    const int hi16   = lane >> 4;
    const int aoff   = hi16 ? 8 : 0;
    const int boff   = hi16 ? 8 : 0;
    const int r0     = wv * 16;          // strip base row in window
    const int rA     = r0 + rl;          // A-operand row for this lane
    const int mrow   = hi16 * 8;         // C/D row offset for this lane

    // ---------------- GEMM1: S = Q K^T (16x128 strip per wave) ----------------
    v16h aq[2];
#pragma unroll
    for (int ks = 0; ks < 2; ++ks) {
        const _Float16* pa = sQ + rA * HD + ks * 32 + aoff;
        aq[ks] = make_frag(pa, pa + 16);
    }
    v8f acc[8];
#pragma unroll
    for (int tc = 0; tc < 8; ++tc) {
        v8f c = {};
#pragma unroll
        for (int ks = 0; ks < 2; ++ks) {
            const _Float16* pb = sK + (tc * 16 + rl) * HD + ks * 32 + boff;
            v16h bf = make_frag(pb, pb + 16);
            c = __builtin_amdgcn_wmma_f32_16x16x32_f16(
                    false, aq[ks], false, bf, (short)0, c, false, false);
        }
        acc[tc] = c;
    }

    // ---------------- scale + rel-pos bias + shift mask + softmax --------------
    const bool masked_win = (w == NW - 1);
#pragma unroll
    for (int m = 0; m < 8; ++m) {
        const int i = r0 + m + mrow;
#pragma unroll
        for (int tc = 0; tc < 8; ++tc) {
            const int j = tc * 16 + rl;
            float val = acc[tc][m] * QKSCALE + sBias[i - j + (WSZ - 1)];
            if (masked_win && ((i < SHIFT) != (j < SHIFT))) val -= 100.0f;
            acc[tc][m] = val;
        }
    }
#pragma unroll
    for (int m = 0; m < 8; ++m) {
        float mx = -3.0e38f;
#pragma unroll
        for (int tc = 0; tc < 8; ++tc) mx = fmaxf(mx, acc[tc][m]);
#pragma unroll
        for (int s = 1; s < 16; s <<= 1) mx = fmaxf(mx, __shfl_xor(mx, s, 32));
        float sum = 0.0f;
#pragma unroll
        for (int tc = 0; tc < 8; ++tc) {
            float e = __expf(acc[tc][m] - mx);
            acc[tc][m] = e;
            sum += e;
        }
#pragma unroll
        for (int s = 1; s < 16; s <<= 1) sum += __shfl_xor(sum, s, 32);
        const float inv = 1.0f / sum;
#pragma unroll
        for (int tc = 0; tc < 8; ++tc) acc[tc][m] *= inv;
    }

    // ---------------- emit attn output (second tuple element) -----------------
    // 134 MB pure stream, never re-read -> non-temporal stores (keep L2 clean).
    float* attn_w = attn_out + (size_t)widx * (WSZ * WSZ);
#pragma unroll
    for (int m = 0; m < 8; ++m) {
        const int i = r0 + m + mrow;
#pragma unroll
        for (int tc = 0; tc < 8; ++tc)
            __builtin_nontemporal_store(acc[tc][m], attn_w + i * WSZ + tc * 16 + rl);
    }

    __syncthreads();  // all waves finished reading sQ/sK; safe to alias with sP
#pragma unroll
    for (int m = 0; m < 8; ++m) {
        const int i = r0 + m + mrow;
#pragma unroll
        for (int tc = 0; tc < 8; ++tc)
            sP[i * WSZ + tc * 16 + rl] = (_Float16)acc[tc][m];
    }
    __syncthreads();

    // ---------------- GEMM2: O = P V (16x64 strip per wave) -------------------
    v8f oacc[4] = {v8f{}, v8f{}, v8f{}, v8f{}};
#pragma unroll
    for (int ks = 0; ks < 4; ++ks) {
        const _Float16* pa = sP + rA * WSZ + ks * 32 + aoff;
        v16h af = make_frag(pa, pa + 16);
#pragma unroll
        for (int vc = 0; vc < 4; ++vc) {
            const _Float16* pb = sVt + (vc * 16 + rl) * WSZ + ks * 32 + boff;
            v16h bf = make_frag(pb, pb + 16);
            oacc[vc] = __builtin_amdgcn_wmma_f32_16x16x32_f16(
                           false, af, false, bf, (short)0, oacc[vc], false, false);
        }
    }

    // ---------------- write x with reverse cyclic shift (roll +64) -------------
#pragma unroll
    for (int m = 0; m < 8; ++m) {
        const int i    = r0 + m + mrow;
        const int grow = (w * WSZ + i + SHIFT) & (WTOT - 1);
        float* xo = xout + ((size_t)b * WTOT + grow) * HD;
#pragma unroll
        for (int vc = 0; vc < 4; ++vc)
            __builtin_nontemporal_store(oacc[vc][m], xo + vc * 16 + rl);
    }
}

extern "C" void kernel_launch(void* const* d_in, const int* in_sizes, int n_in,
                              void* d_out, int out_size, void* d_ws, size_t ws_size,
                              hipStream_t stream)
{
    const float* q   = (const float*)d_in[0];
    const float* k   = (const float*)d_in[1];
    const float* v   = (const float*)d_in[2];
    const float* tbl = (const float*)d_in[3];

    const int B = in_sizes[0] / (WTOT * HD);   // 32
    float* xout = (float*)d_out;                              // (B, W, C)
    float* attn = xout + (size_t)B * WTOT * HD;               // (B*nw, ws, ws)

    dim3 grid(B * NW), block(256);
    hipLaunchKernelGGL(swin_attn_fwd, grid, block, 0, stream,
                       q, k, v, tbl, xout, attn);
}
